// NunchakuSDXLSelfAttention_3186865734262
// MI455X (gfx1250) — 16-bit and 8-bit matrix operand order repaired and checked by reading, compiled, not run on hardware
//
#include <hip/hip_runtime.h>
#include <hip/hip_bf16.h>

typedef __attribute__((ext_vector_type(16))) _Float16 v16h;
typedef __attribute__((ext_vector_type(8)))  _Float16 v8h;
typedef __attribute__((ext_vector_type(8)))  float    v8f;
typedef __attribute__((ext_vector_type(8)))  int      v8i;

__device__ __forceinline__ float rmax16(float v) {
  v = fmaxf(v, __shfl_xor(v, 1, 32));
  v = fmaxf(v, __shfl_xor(v, 2, 32));
  v = fmaxf(v, __shfl_xor(v, 4, 32));
  v = fmaxf(v, __shfl_xor(v, 8, 32));
  return v;
}
__device__ __forceinline__ float rsum16(float v) {
  v += __shfl_xor(v, 1, 32);
  v += __shfl_xor(v, 2, 32);
  v += __shfl_xor(v, 4, 32);
  v += __shfl_xor(v, 8, 32);
  return v;
}
__device__ __forceinline__ v16h ld32h(const _Float16* p) {
  v8h lo = *(const v8h*)p;
  v8h hi = *(const v8h*)(p + 8);
  return __builtin_shufflevector(lo, hi, 0,1,2,3,4,5,6,7,8,9,10,11,12,13,14,15);
}
__device__ __forceinline__ void async_g2l_b128(const void* gaddr, void* lds) {
  unsigned l = (unsigned)(uintptr_t)lds;
  unsigned long long g = (unsigned long long)(uintptr_t)gaddr;
  __asm__ volatile("global_load_async_to_lds_b128 %0, %1, off"
                   :: "v"(l), "v"(g) : "memory");
}
__device__ __forceinline__ void wait_async0() {
  __asm__ volatile("s_wait_asynccnt 0x0" ::: "memory");
}

__global__ void __launch_bounds__(256)
fakequant4_kernel(const float* __restrict__ src, signed char* __restrict__ dst,
                  float* __restrict__ scales, long total_groups)
{
  long gid = (long)blockIdx.x * 8 + (threadIdx.x >> 5);
  int lane = threadIdx.x & 31;
  if (gid >= total_groups) return;
  size_t base = (size_t)gid * 64;
  float a = src[base + lane];
  float b = src[base + 32 + lane];
  float m = fmaxf(fabsf(a), fabsf(b));
  m = fmaxf(m, __shfl_xor(m, 16, 32));
  m = rmax16(m);
  float scale = fmaxf(m * (1.0f / 7.0f), 1e-8f);
  float inv = 1.0f / scale;
  int qa = __float2int_rn(a * inv);
  int qb = __float2int_rn(b * inv);
  qa = qa < -8 ? -8 : (qa > 7 ? 7 : qa);
  qb = qb < -8 ? -8 : (qb > 7 ? 7 : qb);
  dst[base + lane]      = (signed char)qa;
  dst[base + 32 + lane] = (signed char)qb;
  if (lane == 0) scales[gid] = scale;
}

__global__ void __launch_bounds__(256)
lowrank_down_kernel(const float* __restrict__ x, const float* __restrict__ down,
                    float* __restrict__ t, int M, int K)
{
  int row = blockIdx.x * 8 + (threadIdx.x >> 5);
  int r = threadIdx.x & 31;
  if (row >= M) return;
  const float* xr = x + (size_t)row * K;
  float acc = 0.f;
  for (int k = 0; k < K; ++k)
    acc = fmaf(xr[k], down[(size_t)k * 32 + r], acc);
  t[(size_t)row * 32 + r] = acc;
}

__global__ void __launch_bounds__(256)
w4a4_gemm_kernel(const signed char* __restrict__ Aq, const float* __restrict__ As,
                 const signed char* __restrict__ Bq, const float* __restrict__ Bs,
                 const float* __restrict__ T, const float* __restrict__ Up,
                 const float* __restrict__ bias,
                 float* __restrict__ Cf, _Float16* __restrict__ Ch,
                 int M, int N, int K)
{
  __shared__ signed char lA[2][16 * 64];
  __shared__ float lAs[2][16];
  __shared__ float lT[16 * 32];

  const int ng   = K >> 6;
  const int tid  = threadIdx.x;
  const int wave = tid >> 5, lane = tid & 31;
  const int hf   = lane >> 4, l15 = lane & 15;
  const int mbase = blockIdx.x * 16;
  const int nbase = blockIdx.y * 128 + wave * 16;
  const int ncol  = nbase + l15;

  for (int i = tid; i < 16 * 32; i += 256)
    lT[i] = T[(size_t)(mbase + (i >> 5)) * 32 + (i & 31)];

  auto stageA = [&](int g, int buf) {
    if (tid < 64) {
      int r = tid >> 2, c = (tid & 3) << 4;
      async_g2l_b128(Aq + (size_t)(mbase + r) * K + (size_t)g * 64 + c,
                     &lA[buf][r * 64 + c]);
    }
    if (tid < 16) lAs[buf][tid] = As[(size_t)(mbase + tid) * ng + g];
  };

  float acc[8] = {0.f, 0.f, 0.f, 0.f, 0.f, 0.f, 0.f, 0.f};

  stageA(0, 0);
  wait_async0();
  __syncthreads();

  for (int g = 0; g < ng; ++g) {
    const int buf = g & 1;
    if (g + 1 < ng) stageA(g + 1, buf ^ 1);

    const int* ar = (const int*)(&lA[buf][l15 * 64]);
    int ao = hf * 4;
    v8i afr = {ar[ao], ar[ao + 1], ar[ao + 2],  ar[ao + 3],
               ar[ao + 8], ar[ao + 9], ar[ao + 10], ar[ao + 11]};

    const signed char* bp = Bq + (size_t)ncol * K + (size_t)g * 64 + hf * 16;
    int4 b0 = *(const int4*)bp;
    int4 b1 = *(const int4*)(bp + 32);
    v8i bfr = {b0.x, b0.y, b0.z, b0.w, b1.x, b1.y, b1.z, b1.w};

    v8i ic = {};
    ic = __builtin_amdgcn_wmma_i32_16x16x64_iu8(true, afr, true, bfr, ic,
                                                false, false);

    float sw = Bs[(size_t)ncol * ng + g];
#pragma unroll
    for (int j = 0; j < 8; ++j)
      acc[j] += (float)ic[j] * lAs[buf][j + 8 * hf] * sw;

    wait_async0();
    __syncthreads();
  }

  const float* upc = Up + ncol;
#pragma unroll 4
  for (int r = 0; r < 32; ++r) {
    float uv = upc[(size_t)r * N];
#pragma unroll
    for (int j = 0; j < 8; ++j)
      acc[j] += lT[(j + 8 * hf) * 32 + r] * uv;
  }
  float bv = bias ? bias[ncol] : 0.f;
  if (Ch) {
#pragma unroll
    for (int j = 0; j < 8; ++j)
      Ch[(size_t)(mbase + j + 8 * hf) * N + ncol] = (_Float16)(acc[j] + bv);
  } else {
#pragma unroll
    for (int j = 0; j < 8; ++j)
      Cf[(size_t)(mbase + j + 8 * hf) * N + ncol] = acc[j] + bv;
  }
}

#define ATT_S  1024
#define ATT_D3 3840
__global__ void __launch_bounds__(256)
attn_kernel(const _Float16* __restrict__ qkv, float* __restrict__ out)
{
  __shared__ _Float16 lK[32 * 72];
  __shared__ _Float16 lV[64 * 40];
  __shared__ _Float16 lP[8 * 16 * 40];

  const int tid  = threadIdx.x;
  const int wave = tid >> 5, lane = tid & 31;
  const int hf   = lane >> 4, l15 = lane & 15;
  const int b = blockIdx.z, h = blockIdx.y;
  const int qbase = blockIdx.x * 128 + wave * 16;
  const _Float16* qrow =
      qkv + (size_t)(b * ATT_S + qbase + l15) * ATT_D3 + h * 64;

  v16h aQ0, aQ1;
#pragma unroll
  for (int e = 0; e < 16; ++e) {
    int d = hf * 16 + e;
    aQ0[e] = (_Float16)((float)qrow[d] * 0.125f);
    aQ1[e] = (_Float16)((float)qrow[d + 32] * 0.125f);
  }

  float rmx[8], rsm[8];
  v8f o0 = {}, o1 = {}, o2 = {}, o3 = {};
#pragma unroll
  for (int j = 0; j < 8; ++j) { rmx[j] = -1e30f; rsm[j] = 0.f; }

  _Float16* myP = lP + wave * (16 * 40);
  const int key8 = tid >> 3, ch8 = tid & 7;

  for (int kb = 0; kb < ATT_S; kb += 32) {
    __syncthreads();
    {
      const size_t krow = (size_t)(b * ATT_S + kb) * ATT_D3 + h * 64;
      async_g2l_b128(qkv + krow + (size_t)key8 * ATT_D3 + 1280 + ch8 * 8,
                     lK + key8 * 72 + ch8 * 8);
      v8h vv = *(const v8h*)(qkv + krow + (size_t)key8 * ATT_D3 + 2560 + ch8 * 8);
#pragma unroll
      for (int e = 0; e < 8; ++e)
        lV[(ch8 * 8 + e) * 40 + key8] = vv[e];
    }
    wait_async0();
    __syncthreads();

    v8f s0 = {}, s1 = {};
    v16h bk;
    bk = ld32h(lK + l15 * 72 + hf * 16);
    s0 = __builtin_amdgcn_wmma_f32_16x16x32_f16(false, aQ0, false, bk, (short)0, s0, false, false);
    bk = ld32h(lK + l15 * 72 + 32 + hf * 16);
    s0 = __builtin_amdgcn_wmma_f32_16x16x32_f16(false, aQ1, false, bk, (short)0, s0, false, false);
    bk = ld32h(lK + (16 + l15) * 72 + hf * 16);
    s1 = __builtin_amdgcn_wmma_f32_16x16x32_f16(false, aQ0, false, bk, (short)0, s1, false, false);
    bk = ld32h(lK + (16 + l15) * 72 + 32 + hf * 16);
    s1 = __builtin_amdgcn_wmma_f32_16x16x32_f16(false, aQ1, false, bk, (short)0, s1, false, false);

#pragma unroll
    for (int j = 0; j < 8; ++j) {
      float tmax = rmax16(fmaxf(s0[j], s1[j]));
      float mn = fmaxf(rmx[j], tmax);
      float corr = __expf(rmx[j] - mn);
      rmx[j] = mn;
      float p0 = __expf(s0[j] - mn);
      float p1 = __expf(s1[j] - mn);
      rsm[j] = rsm[j] * corr + rsum16(p0 + p1);
      o0[j] *= corr; o1[j] *= corr; o2[j] *= corr; o3[j] *= corr;
      myP[(j + 8 * hf) * 40 + l15]      = (_Float16)p0;
      myP[(j + 8 * hf) * 40 + 16 + l15] = (_Float16)p1;
    }
    __asm__ volatile("s_wait_dscnt 0x0" ::: "memory");

    v8h plo = *(const v8h*)(myP + l15 * 40 + hf * 16);
    v8h phi = *(const v8h*)(myP + l15 * 40 + hf * 16 + 8);
    v16h pa = __builtin_shufflevector(plo, phi, 0,1,2,3,4,5,6,7,8,9,10,11,12,13,14,15);

    v16h bv;
    bv = ld32h(lV + (0 * 16 + l15) * 40 + hf * 16);
    o0 = __builtin_amdgcn_wmma_f32_16x16x32_f16(false, pa, false, bv, (short)0, o0, false, false);
    bv = ld32h(lV + (1 * 16 + l15) * 40 + hf * 16);
    o1 = __builtin_amdgcn_wmma_f32_16x16x32_f16(false, pa, false, bv, (short)0, o1, false, false);
    bv = ld32h(lV + (2 * 16 + l15) * 40 + hf * 16);
    o2 = __builtin_amdgcn_wmma_f32_16x16x32_f16(false, pa, false, bv, (short)0, o2, false, false);
    bv = ld32h(lV + (3 * 16 + l15) * 40 + hf * 16);
    o3 = __builtin_amdgcn_wmma_f32_16x16x32_f16(false, pa, false, bv, (short)0, o3, false, false);
  }

#pragma unroll
  for (int j = 0; j < 8; ++j) {
    int m = j + 8 * hf;
    float inv = 1.0f / (rsm[j] + 1e-30f);
    size_t r = (size_t)(b * ATT_S + qbase + m) * 1280 + h * 64;
    out[r + 0  + l15] = o0[j] * inv;
    out[r + 16 + l15] = o1[j] * inv;
    out[r + 32 + l15] = o2[j] * inv;
    out[r + 48 + l15] = o3[j] * inv;
  }
}

extern "C" void kernel_launch(void* const* d_in, const int* in_sizes, int n_in,
                              void* d_out, int out_size, void* d_ws, size_t ws_size,
                              hipStream_t stream)
{
  const float* x        = (const float*)d_in[0];
  const float* qkv_w    = (const float*)d_in[1];
  const float* qkv_down = (const float*)d_in[2];
  const float* qkv_up   = (const float*)d_in[3];
  const float* out_w    = (const float*)d_in[4];
  const float* out_b    = (const float*)d_in[5];
  const float* out_down = (const float*)d_in[6];
  const float* out_up   = (const float*)d_in[7];
  float* out = (float*)d_out;

  const int M = 8 * 1024, D = 1280, N3 = 3840, NG = D / 64;

  char* w = (char*)d_ws;
  auto take = [&](size_t bytes) {
    char* p = w;
    w += (bytes + 255) & ~(size_t)255;
    return p;
  };
  signed char* xq  = (signed char*)take((size_t)M * D);
  float* xs        = (float*)take((size_t)M * NG * 4);
  signed char* wq1 = (signed char*)take((size_t)N3 * D);
  float* ws1       = (float*)take((size_t)N3 * NG * 4);
  signed char* wq2 = (signed char*)take((size_t)D * D);
  float* ws2       = (float*)take((size_t)D * NG * 4);
  float* t1        = (float*)take((size_t)M * 32 * 4);
  float* t2        = (float*)take((size_t)M * 32 * 4);
  _Float16* qkvb   = (_Float16*)take((size_t)M * N3 * 2);
  float* attno     = (float*)take((size_t)M * D * 4);
  signed char* aq  = (signed char*)take((size_t)M * D);
  float* as2       = (float*)take((size_t)M * NG * 4);

  const long gx = (long)M * NG, gw1 = (long)N3 * NG, gw2 = (long)D * NG;

  fakequant4_kernel<<<dim3((unsigned)((gx  + 7) / 8)), dim3(256), 0, stream>>>(x, xq, xs, gx);
  fakequant4_kernel<<<dim3((unsigned)((gw1 + 7) / 8)), dim3(256), 0, stream>>>(qkv_w, wq1, ws1, gw1);
  fakequant4_kernel<<<dim3((unsigned)((gw2 + 7) / 8)), dim3(256), 0, stream>>>(out_w, wq2, ws2, gw2);
  lowrank_down_kernel<<<dim3(M / 8), dim3(256), 0, stream>>>(x, qkv_down, t1, M, D);

  w4a4_gemm_kernel<<<dim3(M / 16, N3 / 128), dim3(256), 0, stream>>>(
      xq, xs, wq1, ws1, t1, qkv_up, nullptr, nullptr, qkvb, M, N3, D);

  attn_kernel<<<dim3(8, 20, 8), dim3(256), 0, stream>>>(qkvb, attno);

  fakequant4_kernel<<<dim3((unsigned)((gx + 7) / 8)), dim3(256), 0, stream>>>(attno, aq, as2, gx);
  lowrank_down_kernel<<<dim3(M / 8), dim3(256), 0, stream>>>(attno, out_down, t2, M, D);

  w4a4_gemm_kernel<<<dim3(M / 16, D / 128), dim3(256), 0, stream>>>(
      aq, as2, wq2, ws2, t2, out_up, out_b, out, nullptr, M, D, D);
}
